// MANA_75926431859189
// MI455X (gfx1250) — matrix operand order repaired and checked by reading, not compiled, not run on hardware
//
#include <hip/hip_runtime.h>
#include <math.h>

typedef __bf16 bf16;
typedef __attribute__((ext_vector_type(16))) __bf16 bf16x16;
typedef __attribute__((ext_vector_type(8)))  float  f32x8;

#define F   128
#define RBF 20
#define NLAY 3
#define NSTATE 4
#define NP  3

__device__ __forceinline__ float silu_f(float x) { return x / (1.0f + __expf(-x)); }

union Frag16 { bf16x16 v; uint4 q[2]; };

// A fragment 16x32 bf16 from LDS tile (row-major, `stride` bf16 per row).
// lane<16: k = base..base+7 (elems 0..7), 16+base.. (elems 8..15), base = 8*(lane>=16)
__device__ __forceinline__ bf16x16 load_afrag(const bf16* A, int stride, int kt, int lane) {
  int m  = lane & 15;
  int kb = 32 * kt + ((lane >> 4) << 3);
  const bf16* p = A + m * stride + kb;
  Frag16 u;
  u.q[0] = *(const uint4*)(p);       // k = kb .. kb+7
  u.q[1] = *(const uint4*)(p + 16);  // k = kb+16 .. kb+23
  return u.v;
}

// B fragment 32x16 bf16 from global W^T (N x K row-major, strideK bf16 per row).
// lane holds column n = lane&15, k = base..base+7 (elems 0..7), 16+base.. (elems 8..15), base = 8*(lane>=16)
__device__ __forceinline__ bf16x16 load_bfrag(const bf16* WT, int strideK, int nt, int kt, int lane) {
  int n  = nt * 16 + (lane & 15);
  int kb = 32 * kt + ((lane >> 4) << 3);
  const bf16* p = WT + (size_t)n * strideK + kb;
  Frag16 u;
  u.q[0] = *(const uint4*)(p);
  u.q[1] = *(const uint4*)(p + 16);
  return u.v;
}

// ---------------- init / weight prep ----------------
__global__ __launch_bounds__(256)
void init_s_kernel(const int* z, const float* emb, float* s, int Ntot) {
  int i = blockIdx.x * 256 + threadIdx.x;
  if (i < Ntot * F) { int n = i >> 7, f = i & (F - 1); s[i] = emb[(size_t)z[n] * F + f]; }
}

// in: K x Nc f32 row-major; out: Nc x K bf16 row-major (i.e. B^T, contiguous K)
__global__ __launch_bounds__(256)
void transpose_bf16_kernel(const float* in, bf16* out, int K, int Nc) {
  int i = blockIdx.x * 256 + threadIdx.x;
  if (i < K * Nc) {
    int n = i / K, k = i - n * K;
    out[(size_t)n * K + k] = (bf16)in[(size_t)k * Nc + n];
  }
}

// ---------------- per-layer edge message kernel ----------------
__global__ __launch_bounds__(256)
void edge_msg_kernel(const int* ei, const float* pos, const float* centers, const float* gam,
                     const float* fw1, const float* fb1, const bf16* fw2T, const float* fb2,
                     const float* s, const float* v, float* m_s, float* m_v, int Etot) {
  __shared__ __align__(16) float sh_dir[16][3];
  __shared__ __align__(16) float sh_rbf[16][RBF];
  __shared__ int   sh_row[16];
  __shared__ int   sh_col[16];
  __shared__ __align__(16) bf16  sh_h1[16][F];
  __shared__ __align__(16) float sh_sc[16][F];
  __shared__ __align__(16) float sh_phi[16][3 * F];

  int t  = threadIdx.x;
  int e0 = blockIdx.x * 16;

  if (t < 16) {
    int e = e0 + t; if (e >= Etot) e = Etot - 1;
    int r = ei[e], c = ei[Etot + e];
    sh_row[t] = r; sh_col[t] = c;
    float dx = pos[c * 3 + 0] - pos[r * 3 + 0];
    float dy = pos[c * 3 + 1] - pos[r * 3 + 1];
    float dz = pos[c * 3 + 2] - pos[r * 3 + 2];
    float dist = sqrtf(dx * dx + dy * dy + dz * dz + 1e-12f);
    float inv  = 1.0f / (dist + 1e-8f);
    sh_dir[t][0] = dx * inv; sh_dir[t][1] = dy * inv; sh_dir[t][2] = dz * inv;
    for (int rr = 0; rr < RBF; ++rr) {
      float d = dist - centers[rr];
      sh_rbf[t][rr] = __expf(-gam[rr] * d * d);
    }
  }
  __syncthreads();

  // gather sc = s[col] and compute h1 = silu(rbf @ fw1 + fb1)
  for (int i = t; i < 16 * F; i += 256) {
    int e = i >> 7, f = i & (F - 1);
    sh_sc[e][f] = s[(size_t)sh_col[e] * F + f];
  }
  for (int i = t; i < 16 * F; i += 256) {
    int e = i >> 7, f = i & (F - 1);
    float acc = fb1[f];
#pragma unroll
    for (int rr = 0; rr < RBF; ++rr) acc += sh_rbf[e][rr] * fw1[rr * F + f];
    sh_h1[e][f] = (bf16)silu_f(acc);
  }
  __syncthreads();

  // phi = h1 @ fw2 + fb2 via WMMA (16 edges x 384 cols, K=128)
  int lane = t & 31, wave = t >> 5;
  for (int ti = 0; ti < 3; ++ti) {
    int nt = wave + ti * 8;              // 0..23 column tiles of 16
    f32x8 acc = {};
#pragma unroll
    for (int kt = 0; kt < 4; ++kt) {
      bf16x16 a = load_afrag(&sh_h1[0][0], F, kt, lane);
      bf16x16 b = load_bfrag(fw2T, F, nt, kt, lane);
      acc = __builtin_amdgcn_wmma_f32_16x16x32_bf16(false, a, false, b, (short)0, acc, false, false);
    }
    int n = nt * 16 + (lane & 15);
    float bias = fb2[n];
    int mb = (lane >> 4) * 8;
#pragma unroll
    for (int j = 0; j < 8; ++j) sh_phi[mb + j][n] = acc[j] + bias;
  }
  __syncthreads();

  // scatter messages with atomics
  for (int i = t; i < 16 * F; i += 256) {
    int e = i >> 7, f = i & (F - 1);
    if (e0 + e >= Etot) continue;
    int rdst = sh_row[e], csrc = sh_col[e];
    float pss = sh_phi[e][f];
    float pvv = sh_phi[e][F + f];
    float psv = sh_phi[e][2 * F + f];
    float sc  = sh_sc[e][f];
    atomicAdd(&m_s[(size_t)rdst * F + f], pss * sc);
    size_t vb = ((size_t)csrc * F + f) * 3;
    size_t ob = ((size_t)rdst * F + f) * 3;
    float svc = psv * sc;
#pragma unroll
    for (int k = 0; k < 3; ++k) {
      float mv = pvv * v[vb + k] + svc * sh_dir[e][k];
      atomicAdd(&m_v[ob + k], mv);
    }
  }
}

// ---------------- per-layer node update kernel ----------------
__global__ __launch_bounds__(256)
void node_update_kernel(const bf16* uw1T, const float* ub1, const bf16* uw2T, const float* ub2,
                        float* s, float* v, const float* m_s, const float* m_v, int Ntot) {
  __shared__ __align__(16) bf16  sh_x[16][3 * F];
  __shared__ __align__(16) bf16  sh_u1[16][F];
  __shared__ __align__(16) float sh_upd[16][3 * F];
  int t  = threadIdx.x;
  int n0 = blockIdx.x * 16;

  for (int i = t; i < 16 * 3 * F; i += 256) {
    int r = i / (3 * F), c = i - r * (3 * F);
    int node = n0 + r; if (node >= Ntot) node = Ntot - 1;
    float val;
    if (c < F)            val = s[(size_t)node * F + c];
    else if (c < 2 * F)   val = m_s[(size_t)node * F + (c - F)];
    else {
      size_t b = ((size_t)node * F + (c - 2 * F)) * 3;
      float x0 = m_v[b], x1 = m_v[b + 1], x2 = m_v[b + 2];
      val = sqrtf(x0 * x0 + x1 * x1 + x2 * x2 + 1e-12f);
    }
    sh_x[r][c] = (bf16)val;
  }
  __syncthreads();

  int lane = t & 31, wave = t >> 5;
  {  // u1 = silu(x @ uw1 + ub1): 16x384 @ 384x128
    int nt = wave;                       // 8 tiles of 16 cols
    f32x8 acc = {};
#pragma unroll
    for (int kt = 0; kt < 12; ++kt) {
      bf16x16 a = load_afrag(&sh_x[0][0], 3 * F, kt, lane);
      bf16x16 b = load_bfrag(uw1T, 3 * F, nt, kt, lane);
      acc = __builtin_amdgcn_wmma_f32_16x16x32_bf16(false, a, false, b, (short)0, acc, false, false);
    }
    int n = nt * 16 + (lane & 15);
    float bias = ub1[n];
    int mb = (lane >> 4) * 8;
#pragma unroll
    for (int j = 0; j < 8; ++j) sh_u1[mb + j][n] = (bf16)silu_f(acc[j] + bias);
  }
  __syncthreads();

  for (int ti = 0; ti < 3; ++ti) {  // upd = u1 @ uw2 + ub2: 16x128 @ 128x384
    int nt = wave + ti * 8;
    f32x8 acc = {};
#pragma unroll
    for (int kt = 0; kt < 4; ++kt) {
      bf16x16 a = load_afrag(&sh_u1[0][0], F, kt, lane);
      bf16x16 b = load_bfrag(uw2T, F, nt, kt, lane);
      acc = __builtin_amdgcn_wmma_f32_16x16x32_bf16(false, a, false, b, (short)0, acc, false, false);
    }
    int n = nt * 16 + (lane & 15);
    float bias = ub2[n];
    int mb = (lane >> 4) * 8;
#pragma unroll
    for (int j = 0; j < 8; ++j) sh_upd[mb + j][n] = acc[j] + bias;
  }
  __syncthreads();

  for (int i = t; i < 16 * F; i += 256) {
    int r = i >> 7, f = i & (F - 1);
    int node = n0 + r;
    if (node >= Ntot) continue;
    float ds = sh_upd[r][f];
    float al = sh_upd[r][F + f];
    float be = sh_upd[r][2 * F + f];
    s[(size_t)node * F + f] += ds;
    size_t b = ((size_t)node * F + f) * 3;
#pragma unroll
    for (int k = 0; k < 3; ++k) v[b + k] = al * v[b + k] + be * m_v[b + k];
  }
}

// ---------------- heads ----------------
__global__ __launch_bounds__(128)
void energy_kernel(const float* s, const float* ew1, const float* eb1, const float* ew2,
                   const float* eb2, const int* batch, float* out_E) {
  int node = blockIdx.x;
  int st   = blockIdx.y;
  int g    = threadIdx.x;
  __shared__ float sh_s[F];
  __shared__ float red[F];
  sh_s[g] = s[(size_t)node * F + g];
  __syncthreads();
  const float* W = ew1 + (size_t)st * F * F;
  float acc = eb1[st * F + g];
  for (int f = 0; f < F; ++f) acc += sh_s[f] * W[f * F + g];
  red[g] = silu_f(acc) * ew2[st * F + g];
  __syncthreads();
  for (int off = 64; off > 0; off >>= 1) {
    if (g < off) red[g] += red[g + off];
    __syncthreads();
  }
  if (g == 0) atomicAdd(&out_E[(size_t)batch[node] * NSTATE + st], red[0] + eb2[st]);
}

__global__ __launch_bounds__(128)
void vec_head_kernel(const float* v, const float* dipole_w, const float* nac_w,
                     const int* batch, float* out_dip, float* out_nac) {
  int node = blockIdx.x;
  int f    = threadIdx.x;
  __shared__ float red[128][12];
  size_t b = ((size_t)node * F + f) * 3;
  float vx = v[b], vy = v[b + 1], vz = v[b + 2];
  float dw = dipole_w[f];
  red[f][0] = vx * dw; red[f][1] = vy * dw; red[f][2] = vz * dw;
#pragma unroll
  for (int p = 0; p < NP; ++p) {
    float nw = nac_w[p * F + f];
    red[f][3 + p * 3 + 0] = vx * nw;
    red[f][3 + p * 3 + 1] = vy * nw;
    red[f][3 + p * 3 + 2] = vz * nw;
  }
  __syncthreads();
  for (int off = 64; off > 0; off >>= 1) {
    if (f < off) {
      for (int j = 0; j < 12; ++j) red[f][j] += red[f + off][j];
    }
    __syncthreads();
  }
  if (f < 12) {
    int m = batch[node];
    if (f < 3) atomicAdd(&out_dip[m * 3 + f], red[0][f]);
    else       atomicAdd(&out_nac[m * 9 + (f - 3)], red[0][f]);
  }
}

__global__ __launch_bounds__(128)
void mol_head_kernel(const float* out_E, const float* out_nac,
                     const float* aw1, const float* ab1, const float* aw2, const float* ab2,
                     const float* yw1, const float* yb1, const float* yw2, const float* yb2,
                     float* out_lam, float* out_phi) {
  int m = blockIdx.x;
  int t = threadIdx.x;
  __shared__ float cat6[6];
  __shared__ float red[128];
  if (t < 3) {
    cat6[t] = out_E[m * NSTATE + 1 + t] - out_E[m * NSTATE];
    const float* nv = out_nac + m * 9 + t * 3;
    cat6[3 + t] = sqrtf(nv[0] * nv[0] + nv[1] * nv[1] + nv[2] * nv[2] + 1e-12f);
  }
  __syncthreads();
  float val = 0.0f;
  if (t < 64) {
    float a = ab1[t];
    for (int i = 0; i < 3; ++i) a += cat6[i] * aw1[i * 64 + t];
    val = silu_f(a) * aw2[t];
  }
  red[t] = val;
  __syncthreads();
  for (int off = 64; off > 0; off >>= 1) { if (t < off) red[t] += red[t + off]; __syncthreads(); }
  if (t == 0) out_lam[m] = red[0] + ab2[0];
  __syncthreads();
  float a = yb1[t];
  for (int i = 0; i < 6; ++i) a += cat6[i] * yw1[i * 128 + t];
  red[t] = silu_f(a) * yw2[t];
  __syncthreads();
  for (int off = 64; off > 0; off >>= 1) { if (t < off) red[t] += red[t + off]; __syncthreads(); }
  if (t == 0) out_phi[m] = 1.0f / (1.0f + __expf(-(red[0] + yb2[0])));
}

// ---------------- launch ----------------
extern "C" void kernel_launch(void* const* d_in, const int* in_sizes, int n_in,
                              void* d_out, int out_size, void* d_ws, size_t ws_size,
                              hipStream_t stream) {
  const int*   z        = (const int*)d_in[0];
  const float* pos      = (const float*)d_in[1];
  const int*   ei       = (const int*)d_in[2];
  const int*   batch    = (const int*)d_in[3];
  const float* emb      = (const float*)d_in[5];
  const float* centers  = (const float*)d_in[6];
  const float* gam      = (const float*)d_in[7];
  const float* fw1      = (const float*)d_in[8];
  const float* fb1      = (const float*)d_in[9];
  const float* fw2      = (const float*)d_in[10];
  const float* fb2      = (const float*)d_in[11];
  const float* uw1      = (const float*)d_in[12];
  const float* ub1      = (const float*)d_in[13];
  const float* uw2      = (const float*)d_in[14];
  const float* ub2      = (const float*)d_in[15];
  const float* ew1      = (const float*)d_in[16];
  const float* eb1      = (const float*)d_in[17];
  const float* ew2      = (const float*)d_in[18];
  const float* eb2      = (const float*)d_in[19];
  const float* dipole_w = (const float*)d_in[20];
  const float* nac_w    = (const float*)d_in[21];
  const float* aw1      = (const float*)d_in[22];
  const float* ab1      = (const float*)d_in[23];
  const float* aw2      = (const float*)d_in[24];
  const float* ab2      = (const float*)d_in[25];
  const float* yw1      = (const float*)d_in[26];
  const float* yb1      = (const float*)d_in[27];
  const float* yw2      = (const float*)d_in[28];
  const float* yb2      = (const float*)d_in[29];

  const int N = in_sizes[0];
  const int E = in_sizes[2] / 2;
  const int M = out_size / 18;  // M*(S + 3 + P*3 + 1 + 1)

  // workspace partition
  char* w = (char*)d_ws;
  float* s_buf = (float*)w;  w += (size_t)N * F * 4;
  float* v_buf = (float*)w;  w += (size_t)N * F * 3 * 4;
  float* m_s   = (float*)w;  w += (size_t)N * F * 4;
  float* m_v   = (float*)w;  w += (size_t)N * F * 3 * 4;
  bf16* fw2T   = (bf16*)w;   w += (size_t)NLAY * 384 * 128 * 2;
  bf16* uw1T   = (bf16*)w;   w += (size_t)NLAY * 128 * 384 * 2;
  bf16* uw2T   = (bf16*)w;   w += (size_t)NLAY * 384 * 128 * 2;
  (void)ws_size; (void)n_in;

  float* out_E   = (float*)d_out;
  float* out_dip = out_E + (size_t)M * NSTATE;
  float* out_nac = out_dip + (size_t)M * 3;
  float* out_lam = out_nac + (size_t)M * NP * 3;
  float* out_phi = out_lam + M;

  hipMemsetAsync(d_out, 0, (size_t)out_size * 4, stream);
  hipMemsetAsync(v_buf, 0, (size_t)N * F * 3 * 4, stream);

  init_s_kernel<<<(N * F + 255) / 256, 256, 0, stream>>>(z, emb, s_buf, N);

  const int TW = 128 * 384;
  for (int l = 0; l < NLAY; ++l) {
    transpose_bf16_kernel<<<(TW + 255) / 256, 256, 0, stream>>>(
        fw2 + (size_t)l * TW, fw2T + (size_t)l * TW, 128, 384);
    transpose_bf16_kernel<<<(TW + 255) / 256, 256, 0, stream>>>(
        uw1 + (size_t)l * TW, uw1T + (size_t)l * TW, 384, 128);
    transpose_bf16_kernel<<<(TW + 255) / 256, 256, 0, stream>>>(
        uw2 + (size_t)l * TW, uw2T + (size_t)l * TW, 128, 384);
  }

  for (int l = 0; l < NLAY; ++l) {
    hipMemsetAsync(m_s, 0, (size_t)N * F * 4, stream);
    hipMemsetAsync(m_v, 0, (size_t)N * F * 3 * 4, stream);
    edge_msg_kernel<<<(E + 15) / 16, 256, 0, stream>>>(
        ei, pos, centers, gam,
        fw1 + (size_t)l * RBF * F, fb1 + (size_t)l * F,
        fw2T + (size_t)l * TW, fb2 + (size_t)l * 384,
        s_buf, v_buf, m_s, m_v, E);
    node_update_kernel<<<(N + 15) / 16, 256, 0, stream>>>(
        uw1T + (size_t)l * TW, ub1 + (size_t)l * F,
        uw2T + (size_t)l * TW, ub2 + (size_t)l * 384,
        s_buf, v_buf, m_s, m_v, N);
  }

  energy_kernel<<<dim3(N, NSTATE), 128, 0, stream>>>(s_buf, ew1, eb1, ew2, eb2, batch, out_E);
  vec_head_kernel<<<N, 128, 0, stream>>>(v_buf, dipole_w, nac_w, batch, out_dip, out_nac);
  mol_head_kernel<<<M, 128, 0, stream>>>(out_E, out_nac, aw1, ab1, aw2, ab2,
                                         yw1, yb1, yw2, yb2, out_lam, out_phi);
}
